// MiniGridGRU_4234837754347
// MI455X (gfx1250) — matrix operand order repaired and checked by reading, not compiled, not run on hardware
//
#include <hip/hip_runtime.h>
#include <hip/hip_bf16.h>
#include <math.h>

// ---------------------------------------------------------------------------
// MiniGridGRU for MI455X (gfx1250, wave32)
//
// Phase 1: igates = xs @ Wi^T + bi as bf16 WMMA GEMM, with LDS tiles staged by
//          the Tensor Data Mover (tensor_load_to_lds + s_wait_tensorcnt),
//          double-buffered so DMA of tile k+1 overlaps WMMA on tile k.
//          ~103 GFLOP / ~176 MB traffic -> memory-bound at 23.3 TB/s; TDM
//          staging cuts intra-block L2 re-reads ~3x vs per-lane global loads.
// Phase 2: sequential GRU scan with Wh pinned in LDS as bf16 (192KB of the
//          320KB WGP LDS), 128 persistent blocks each owning 16 hidden units
//          across all three gates -> one device barrier per step.
// ---------------------------------------------------------------------------

#define SEQ_LEN   4096
#define IN_DIM    2048
#define HID       2048
#define GEMM_N    (3 * HID)          // 6144 gate rows

typedef __attribute__((ext_vector_type(16))) __bf16       v16bf;
typedef __attribute__((ext_vector_type(8)))  float        v8f;
typedef __attribute__((ext_vector_type(4)))  unsigned int v4u;
typedef __attribute__((ext_vector_type(4)))  unsigned int u32x4;
typedef __attribute__((ext_vector_type(8)))  int          i32x8;
typedef __attribute__((ext_vector_type(4)))  int          i32x4;

union Frag16 { v4u u[2]; v16bf v; };   // 32-byte A/B fragment

#if defined(__has_builtin)
#  if __has_builtin(__builtin_amdgcn_tensor_load_to_lds)
#    define HAVE_TDM 1
#  else
#    define HAVE_TDM 0
#  endif
#else
#  define HAVE_TDM 0
#endif

// ------------------------- kernel 1: f32 -> bf16 ---------------------------
__global__ void cvt_f32_bf16(const float* __restrict__ src,
                             __bf16* __restrict__ dst, size_t n) {
  size_t i = (size_t)blockIdx.x * blockDim.x + threadIdx.x;
  size_t stride = (size_t)gridDim.x * blockDim.x;
  for (; i < n; i += stride) dst[i] = (__bf16)src[i];
}

// -------------------- kernel 2: init h buffer + barrier --------------------
__global__ void init_state_kernel(const float* __restrict__ h0,
                                  float* __restrict__ hbufs,
                                  unsigned* __restrict__ counter) {
  int i = blockIdx.x * blockDim.x + threadIdx.x;
  if (i < HID) hbufs[i] = h0[i];          // h buffer 0 = init_state
  if (i == 0) *counter = 0u;              // reset barrier (graph-replay safe)
}

#if HAVE_TDM
// Issue one 2D TDM tile load: global (row-major, stride_elems bf16 per row)
// -> densely packed LDS tile [tile_d1][tile_d0].  D# bitfields per ISA §8.
__device__ __forceinline__ void tdm_load_2d(unsigned lds_off, const void* gptr,
                                            unsigned tensor_d0, unsigned tensor_d1,
                                            unsigned tile_d0, unsigned tile_d1,
                                            unsigned stride_elems) {
  unsigned long long ga = (unsigned long long)(uintptr_t)gptr;
  u32x4 g0;
  g0[0] = 1u;                                          // count=1, user D#
  g0[1] = lds_off;                                     // lds_addr (bytes)
  g0[2] = (unsigned)(ga & 0xFFFFFFFFu);                // global_addr[31:0]
  g0[3] = (unsigned)((ga >> 32) & 0x1FFFFFFu)          // global_addr[56:32]
          | (2u << 30);                                // type=2 ("image")
  i32x8 g1;
  g1[0] = (int)(1u << 16);                             // data_size=1 (2B), mask=0
  g1[1] = (int)((tensor_d0 & 0xFFFFu) << 16);          // tensor_dim0[15:0]
  g1[2] = (int)(((tensor_d0 >> 16) & 0xFFFFu)
          | ((tensor_d1 & 0xFFFFu) << 16));            // dim0[31:16] | dim1[15:0]
  g1[3] = (int)(((tensor_d1 >> 16) & 0xFFFFu)
          | ((tile_d0 & 0xFFFFu) << 16));              // dim1[31:16] | tile_dim0
  g1[4] = (int)(tile_d1 & 0xFFFFu);                    // tile_dim1, tile_dim2=0 (2D)
  g1[5] = (int)stride_elems;                           // tensor_dim0_stride[31:0]
  g1[6] = 0;                                           // stride[47:32], dim1_stride lo
  g1[7] = 0;
  i32x4 z4 = {0, 0, 0, 0};
#if __clang_major__ >= 23
  i32x8 z8 = {0, 0, 0, 0, 0, 0, 0, 0};
  __builtin_amdgcn_tensor_load_to_lds(g0, g1, z4, z4, z8, 0);
#else
  __builtin_amdgcn_tensor_load_to_lds(g0, g1, z4, z4, 0);
#endif
}
#endif

// ------------------- kernel 3: WMMA GEMM for igates ------------------------
// out[t][r] = sum_k X[t][k] * W[r][k] + bi[r]
// Block tile 64(M) x 128(N), K stepped by 64.  A tile (64x64 bf16, 8KB) and
// B tile (128x64 bf16, 16KB) double-buffered in LDS, filled by TDM.
#define BM 64
#define BN 128
#define BK 64
#define KSTEPS (IN_DIM / BK)

__global__ __launch_bounds__(256)
void igates_gemm_wmma(const __bf16* __restrict__ X,   // 4096 x 2048 (row major)
                      const __bf16* __restrict__ W,   // 6144 x 2048 (row major = B^T)
                      const float*  __restrict__ bi,  // 6144
                      float* __restrict__ out)        // 4096 x 6144
{
  __shared__ __bf16 sA[2][BM * BK];   // [m][k], row stride BK
  __shared__ __bf16 sB[2][BN * BK];   // [n][k], row stride BK

  const int lane = threadIdx.x & 31;
  const int wave = threadIdx.x >> 5;      // 0..7
  const int wm   = wave & 1;              // 2 row groups of 32
  const int wn   = wave >> 1;             // 4 col groups of 32
  const int row0 = blockIdx.y * BM;
  const int col0 = blockIdx.x * BN;
  const int lmod = lane & 15;
  const int lhi  = lane >> 4;             // 0 or 1 (lane half)

  // C layout (ISA 7.12.2): vgpr e, lanes0-15 -> (row e,   col lane)
  //                                lanes16-31 -> (row e+8, col lane-16)
  v8f c[2][2];
  #pragma unroll
  for (int i = 0; i < 2; ++i)
    #pragma unroll
    for (int j = 0; j < 2; ++j) {
      float b = bi[col0 + wn * 32 + j * 16 + lmod];
      #pragma unroll
      for (int e = 0; e < 8; ++e) c[i][j][e] = b;   // bias folded into C init
    }

  // ---- tile loaders: TDM (preferred) or cooperative-copy fallback ----
  auto load_tiles = [&](int buf, int k0) {
#if HAVE_TDM
    if (wave == 0) {   // TDM ignores EXEC; issue once per workgroup
      tdm_load_2d((unsigned)(uintptr_t)&sA[buf][0],
                  X + (size_t)row0 * IN_DIM + k0,
                  IN_DIM, SEQ_LEN, BK, BM, IN_DIM);
      tdm_load_2d((unsigned)(uintptr_t)&sB[buf][0],
                  W + (size_t)col0 * IN_DIM + k0,
                  IN_DIM, GEMM_N, BK, BN, IN_DIM);
    }
#else
    // 256 threads copy 64x64 + 128x64 bf16 as b128 chunks (8 bf16 each)
    for (int idx = threadIdx.x; idx < (BM * BK) / 8; idx += 256) {
      int r = idx / (BK / 8), cchunk = idx % (BK / 8);
      *(v4u*)&sA[buf][r * BK + cchunk * 8] =
          *(const v4u*)(X + (size_t)(row0 + r) * IN_DIM + k0 + cchunk * 8);
    }
    for (int idx = threadIdx.x; idx < (BN * BK) / 8; idx += 256) {
      int r = idx / (BK / 8), cchunk = idx % (BK / 8);
      *(v4u*)&sB[buf][r * BK + cchunk * 8] =
          *(const v4u*)(W + (size_t)(col0 + r) * IN_DIM + k0 + cchunk * 8);
    }
#endif
  };
  auto wait_tiles = [&]() {
#if HAVE_TDM
    if (wave == 0) __builtin_amdgcn_s_wait_tensorcnt(0);
#endif
  };

  // Prologue: fill buffer 0
  load_tiles(0, 0);
  wait_tiles();
  __syncthreads();

  for (int ks = 0; ks < KSTEPS; ++ks) {
    const int cur = ks & 1;
    if (ks + 1 < KSTEPS) load_tiles(cur ^ 1, (ks + 1) * BK);  // overlap DMA

    #pragma unroll
    for (int kk = 0; kk < BK; kk += 32) {
      // A fragment 16x32: lane holds row (wm*32+i*16+lmod), K chunks
      // [lhi*8,+8) and [16+lhi*8,+8) of this 32-wide slice (ISA A layout)
      Frag16 a[2];
      #pragma unroll
      for (int i = 0; i < 2; ++i) {
        const __bf16* p = &sA[cur][(wm * 32 + i * 16 + lmod) * BK + kk + lhi * 8];
        a[i].u[0] = *(const v4u*)(p);
        a[i].u[1] = *(const v4u*)(p + 16);
      }
      // B fragment 32x16 (KxN): lane holds column (wn*32+j*16+lmod),
      // K chunks [lhi*8,+8) and [16+lhi*8,+8) (column of B = row of W)
      Frag16 b[2];
      #pragma unroll
      for (int j = 0; j < 2; ++j) {
        const __bf16* p = &sB[cur][(wn * 32 + j * 16 + lmod) * BK + kk + lhi * 8];
        b[j].u[0] = *(const v4u*)(p);
        b[j].u[1] = *(const v4u*)(p + 16);
      }
      #pragma unroll
      for (int i = 0; i < 2; ++i)
        #pragma unroll
        for (int j = 0; j < 2; ++j)
          c[i][j] = __builtin_amdgcn_wmma_f32_16x16x32_bf16(
              /*neg_a=*/false, a[i].v, /*neg_b=*/false, b[j].v,
              /*c_mod=*/(short)0, c[i][j], /*reuse_a=*/false, /*reuse_b=*/false);
    }

    wait_tiles();        // next buffer's DMA complete before anyone reads it
    __syncthreads();
  }

  #pragma unroll
  for (int i = 0; i < 2; ++i)
    #pragma unroll
    for (int j = 0; j < 2; ++j) {
      const int col   = col0 + wn * 32 + j * 16 + lmod;
      const int rbase = row0 + wm * 32 + i * 16 + lhi * 8;
      #pragma unroll
      for (int e = 0; e < 8; ++e)
        out[(size_t)(rbase + e) * GEMM_N + col] = c[i][j][e];
    }
}

// ---------------------- kernel 4: persistent GRU scan ----------------------
#define RBLK      128                 // persistent workgroups (1/WGP)
#define JPB       (HID / RBLK)        // 16 hidden indices per block
#define WH_STRIDE 2056                // padded bf16 LDS row stride (bank spread)

constexpr size_t WH_LDS_BYTES  = (size_t)3 * JPB * WH_STRIDE * sizeof(__bf16); // 197376
constexpr size_t SCAN_LDS_BYTES = WH_LDS_BYTES + HID * sizeof(float)           // h vec
                                  + 3 * JPB * sizeof(float);                   // acc

__global__ __launch_bounds__(256)
void gru_scan(const float* __restrict__ igates,  // 4096 x 6144 [r|z|n]
              const float* __restrict__ Wh,      // 6144 x 2048 f32
              const float* __restrict__ bn,      // 2048
              float* __restrict__ hbufs,         // 2 x 2048 (double buffer)
              unsigned* __restrict__ counter,    // device barrier (monotonic)
              float* __restrict__ out)           // 4096*2048 states + 2048 last
{
  extern __shared__ char smem[];
  __bf16* wh  = (__bf16*)smem;                          // [3][JPB][WH_STRIDE]
  float* hbuf = (float*)(smem + WH_LDS_BYTES);          // [HID]
  float* acc  = hbuf + HID;                             // [3*JPB]

  const int tid = threadIdx.x;
  const int j0  = blockIdx.x * JPB;

  // Pin this block's Wh slice in LDS as bf16 (gate g rows g*HID+j0 .. +JPB)
  for (int idx = tid; idx < 3 * JPB * HID; idx += blockDim.x) {
    int g = idx / (JPB * HID);
    int rem = idx - g * JPB * HID;
    int r = rem / HID;
    int k = rem - r * HID;
    wh[(g * JPB + r) * WH_STRIDE + k] =
        (__bf16)Wh[(size_t)(g * HID + j0 + r) * HID + k];
  }
  const float bnv = (tid < JPB) ? bn[j0 + tid] : 0.0f;

  const int row = tid >> 4;     // 0..15  (dot-product row within slice)
  const int l16 = tid & 15;     // 16 lanes cooperate per row (wave32 halves)

  for (int t = 0; t < SEQ_LEN; ++t) {
    __syncthreads();
    // Stage h_t into LDS (16KB double buffer lives in L2)
    const float* hsrc = hbufs + (size_t)(t & 1) * HID;
    for (int i = tid; i < HID; i += blockDim.x) hbuf[i] = hsrc[i];
    __syncthreads();

    // hr/hz/hn for our 16 hidden indices: 48 dots of length 2048 from LDS
    for (int g = 0; g < 3; ++g) {
      const __bf16* wrow = wh + (g * JPB + row) * WH_STRIDE;
      float p = 0.0f;
      #pragma unroll 8
      for (int k = l16; k < HID; k += 16)
        p = fmaf((float)wrow[k], hbuf[k], p);
      #pragma unroll
      for (int off = 8; off; off >>= 1)       // reduce within 16-lane group
        p += __shfl_xor(p, off, 32);
      if (l16 == 0) acc[g * JPB + row] = p;
    }
    __syncthreads();

    // Elementwise GRU update for our hidden indices
    if (tid < JPB) {
      const int j = j0 + tid;
      const float* ig = igates + (size_t)t * GEMM_N;
      float gr = ig[j], gz = ig[HID + j], gn = ig[2 * HID + j];
      float hr = acc[tid], hz = acc[JPB + tid], hn = acc[2 * JPB + tid];
      float hprev  = hbuf[j];
      float reset  = 1.0f / (1.0f + __expf(-(gr + hr)));
      float update = 1.0f / (1.0f + __expf(-(gz + hz)));
      float nv     = tanhf(gn + reset * (hn + bnv));
      float hnew   = (1.0f - update) * nv + update * hprev;
      hbufs[(size_t)((t + 1) & 1) * HID + j] = hnew;
      out[(size_t)t * HID + j] = hnew;
      if (t == SEQ_LEN - 1) out[(size_t)SEQ_LEN * HID + j] = hnew; // last_state
      if (t + 1 < SEQ_LEN)  // warm L2/L0 for next step's gate row
        __builtin_prefetch((const void*)(igates + (size_t)(t + 1) * GEMM_N + j), 0, 1);
    }

    // One device-wide barrier per step (monotonic counter; reset per launch).
    __threadfence();
    __syncthreads();
    if (tid == 0) {
      __hip_atomic_fetch_add(counter, 1u, __ATOMIC_ACQ_REL,
                             __HIP_MEMORY_SCOPE_AGENT);
      const unsigned target = (unsigned)(t + 1) * RBLK;
      while (__hip_atomic_load(counter, __ATOMIC_ACQUIRE,
                               __HIP_MEMORY_SCOPE_AGENT) < target)
        __builtin_amdgcn_s_sleep(1);
    }
    __syncthreads();
  }
}

// ---------------------------------------------------------------------------
extern "C" void kernel_launch(void* const* d_in, const int* in_sizes, int n_in,
                              void* d_out, int out_size, void* d_ws, size_t ws_size,
                              hipStream_t stream) {
  const float* xs = (const float*)d_in[0];   // 4096 x 2048
  const float* h0 = (const float*)d_in[1];   // 2048
  const float* Wi = (const float*)d_in[2];   // 6144 x 2048
  const float* Wh = (const float*)d_in[3];   // 6144 x 2048
  const float* bi = (const float*)d_in[4];   // 6144
  const float* bn = (const float*)d_in[5];   // 2048
  float* out = (float*)d_out;

  // Workspace layout (~143 MB)
  char* ws = (char*)d_ws;
  size_t o = 0;
  __bf16* xs_bf = (__bf16*)(ws + o); o += (size_t)SEQ_LEN * IN_DIM * sizeof(__bf16);
  __bf16* wi_bf = (__bf16*)(ws + o); o += (size_t)GEMM_N * IN_DIM * sizeof(__bf16);
  float* igates = (float*)(ws + o);  o += (size_t)SEQ_LEN * GEMM_N * sizeof(float);
  float* hbufs  = (float*)(ws + o);  o += (size_t)2 * HID * sizeof(float);
  unsigned* counter = (unsigned*)(ws + o);

  cvt_f32_bf16<<<1024, 256, 0, stream>>>(xs, xs_bf, (size_t)SEQ_LEN * IN_DIM);
  cvt_f32_bf16<<<1024, 256, 0, stream>>>(Wi, wi_bf, (size_t)GEMM_N * IN_DIM);
  init_state_kernel<<<(HID + 255) / 256, 256, 0, stream>>>(h0, hbufs, counter);

  dim3 ggrid(GEMM_N / BN, SEQ_LEN / BM);
  igates_gemm_wmma<<<ggrid, 256, 0, stream>>>(xs_bf, wi_bf, bi, igates);

  gru_scan<<<RBLK, 256, SCAN_LDS_BYTES, stream>>>(igates, Wh, bn, hbufs,
                                                  counter, out);
}
